// SelfAttention1D_57750130262317
// MI455X (gfx1250) — matrix operand order repaired and checked by reading, not compiled, not run on hardware
//
#include <hip/hip_runtime.h>
#include <hip/hip_bf16.h>

typedef __attribute__((ext_vector_type(16))) _Float16 v16h;
typedef __attribute__((ext_vector_type(8)))  _Float16 v8h;
typedef __attribute__((ext_vector_type(8)))  float    v8f;
typedef __attribute__((address_space(3)))    _Float16 lds_f16;

// ---------------------------------------------------------------------------
// Async global->LDS 16B copy (CDNA5 GLOBAL_LOAD_ASYNC_TO_LDS_B128, ASYNCcnt).
// Inline asm per bridge-doc guidance (bypasses toolchain builtin arity skew).
// Pattern: issue per-lane copies -> s_wait_asynccnt 0 -> barrier -> consume.
// ---------------------------------------------------------------------------
__device__ __forceinline__ void cp16_async(_Float16* dst_lds, const _Float16* src) {
  unsigned lds = (unsigned)(size_t)(lds_f16*)dst_lds;
  unsigned long long ga = (unsigned long long)(size_t)src;
  asm volatile("global_load_async_to_lds_b128 %0, %1, off"
               :: "v"(lds), "v"(ga) : "memory");
}
__device__ __forceinline__ void async_wait0() {
  asm volatile("s_wait_asynccnt 0x0" ::: "memory");
}

// ---------------------------------------------------------------------------
// WMMA fragment helpers (layouts per CDNA5 ISA 7.12.2, wave32).
// A 16x32 f16 from LDS tile [m][k] (k contiguous): 2x ds_load_b128.
// B 32x16 f16 from LDS tile [n][k] (k contiguous): 2x ds_load_b128.
// C/D f32: lane L: VGPR r -> M = r + (L&16?8:0), N = L&15.
// ---------------------------------------------------------------------------
__device__ __forceinline__ v16h load_a_frag(const _Float16* src, int ld) {
  int lane = threadIdx.x & 31;
  int m  = lane & 15;
  int kb = (lane & 16) ? 8 : 0;
  const _Float16* r = src + m * ld + kb;
  v8h lo = *(const v8h*)r;
  v8h hi = *(const v8h*)(r + 16);
  return __builtin_shufflevector(lo, hi, 0, 1, 2, 3, 4, 5, 6, 7,
                                 8, 9, 10, 11, 12, 13, 14, 15);
}

__device__ __forceinline__ v16h load_b_frag_nk(const _Float16* src, int ld) {
  int lane = threadIdx.x & 31;
  int n  = lane & 15;
  int kb = (lane & 16) ? 8 : 0;
  const _Float16* r = src + n * ld + kb;
  v8h lo = *(const v8h*)r;
  v8h hi = *(const v8h*)(r + 16);
  return __builtin_shufflevector(lo, hi, 0, 1, 2, 3, 4, 5, 6, 7,
                                 8, 9, 10, 11, 12, 13, 14, 15);
}

__device__ __forceinline__ v8f wmma_f16(v16h a, v16h b, v8f c) {
  return __builtin_amdgcn_wmma_f32_16x16x32_f16(
      false, a, false, b, (short)0, c, false, false);
}

// ---------------------------------------------------------------------------
// GroupNorm -> hT (B, T, C) f16  (transposed so GEMM B-tiles stage as b128)
// ---------------------------------------------------------------------------
__global__ __launch_bounds__(256)
void gn_kernel(const float* __restrict__ x, const float* __restrict__ w,
               const float* __restrict__ b, _Float16* __restrict__ hT,
               int C, int T, int G) {
  int bi = blockIdx.x / G, g = blockIdx.x % G;
  const int cg = 64;                       // C/G
  size_t base = ((size_t)bi * C + (size_t)g * cg) * T;
  int n = cg * T;
  float s = 0.f, s2 = 0.f;
  for (int i = threadIdx.x; i < n; i += 256) {
    float v = x[base + i];
    s += v; s2 += v * v;
  }
  __shared__ float rs[256], rq[256];
  rs[threadIdx.x] = s; rq[threadIdx.x] = s2;
  __syncthreads();
  for (int off = 128; off > 0; off >>= 1) {
    if ((int)threadIdx.x < off) {
      rs[threadIdx.x] += rs[threadIdx.x + off];
      rq[threadIdx.x] += rq[threadIdx.x + off];
    }
    __syncthreads();
  }
  float mean = rs[0] / (float)n;
  float var  = rq[0] / (float)n - mean * mean;
  float rstd = rsqrtf(var + 1e-5f);
  for (int i = threadIdx.x; i < n; i += 256) {
    int c = i & 63, t = i >> 6;
    float v = (x[base + (size_t)c * T + t] - mean) * rstd * w[g * cg + c]
              + b[g * cg + c];
    hT[((size_t)bi * T + t) * C + g * cg + c] = (_Float16)v;
  }
}

__global__ void cvt_kernel(const float* __restrict__ in,
                           _Float16* __restrict__ out, int n) {
  int i = blockIdx.x * 256 + threadIdx.x;
  if (i < n) out[i] = (_Float16)in[i];
}

// ---------------------------------------------------------------------------
// WMMA GEMM: Out[b](MxN) = W(MxK) * XT[b](N,K)^T + bias
//  EPI=0 (QKV): rows<512 get 1/8 softmax scale folded in; rows<1024 -> qkT
//               (B,T,2C) via one v8h store; rows>=1024 -> vN (B,C,T)
//  EPI=1 (proj): f32 out (B,C,T) with residual add
// Block 128 thr = 4 waves; tile 128x64; each wave 32x64 (8 WMMAs / K-step).
// ---------------------------------------------------------------------------
#define GBM 128
#define GBN 64
#define GBK 32
#define GLDA 40
#define GLDB 40

template <int EPI>
__global__ __launch_bounds__(128)
void gemm_kernel(const _Float16* __restrict__ Wm, const _Float16* __restrict__ XT,
                 const float* __restrict__ bias, const float* __restrict__ resid,
                 _Float16* __restrict__ outQK, _Float16* __restrict__ outV,
                 float* __restrict__ outF, int M, int K, int N) {
  __shared__ __align__(16) _Float16 As[GBM * GLDA];
  __shared__ __align__(16) _Float16 Bs[GBN * GLDB];
  int tid = threadIdx.x;
  int wave = tid >> 5, lane = tid & 31;
  int n0 = blockIdx.x * GBN;
  int m0 = blockIdx.y * GBM;
  int bi = blockIdx.z;
  const _Float16* Xb = XT + (size_t)bi * N * K;   // (N,K) layout

  v8f acc[2][4] = {};
  for (int k0 = 0; k0 < K; k0 += GBK) {
    __syncthreads();                               // prior frag reads done
    for (int i = tid; i < GBM * GBK / 8; i += 128) {
      int m = i >> 2, kc = (i & 3) * 8;
      cp16_async(&As[m * GLDA + kc], &Wm[(size_t)(m0 + m) * K + k0 + kc]);
    }
    for (int i = tid; i < GBN * GBK / 8; i += 128) {
      int nn = i >> 2, kc = (i & 3) * 8;
      cp16_async(&Bs[nn * GLDB + kc], &Xb[(size_t)(n0 + nn) * K + k0 + kc]);
    }
    if (k0 + GBK < K) {
      __builtin_prefetch(&Wm[(size_t)m0 * K + k0 + GBK], 0, 1);
      __builtin_prefetch(&Xb[(size_t)n0 * K + k0 + GBK], 0, 1);
    }
    async_wait0();                                 // own wave's copies landed
    __syncthreads();                               // all waves' copies landed
    v16h bf[4];
#pragma unroll
    for (int nt = 0; nt < 4; ++nt)
      bf[nt] = load_b_frag_nk(&Bs[(nt * 16) * GLDB], GLDB);
#pragma unroll
    for (int mt = 0; mt < 2; ++mt) {
      v16h a = load_a_frag(&As[(wave * 32 + mt * 16) * GLDA], GLDA);
#pragma unroll
      for (int nt = 0; nt < 4; ++nt)
        acc[mt][nt] = wmma_f16(a, bf[nt], acc[mt][nt]);
    }
  }

  int n_l = lane & 15, hf = (lane >> 4) & 1;
#pragma unroll
  for (int mt = 0; mt < 2; ++mt) {
    int rowbase = m0 + wave * 32 + mt * 16 + hf * 8;  // 8 consecutive rows/lane
#pragma unroll
    for (int nt = 0; nt < 4; ++nt) {
      int col = n0 + nt * 16 + n_l;
      if (EPI == 0) {
        float sc = (rowbase < 512) ? 0.125f : 1.0f;   // fold q softmax scale
        v8h pk;
#pragma unroll
        for (int r = 0; r < 8; ++r)
          pk[r] = (_Float16)((acc[mt][nt][r] + bias[rowbase + r]) * sc);
        if (rowbase < 1024) {   // q,k -> transposed (B,T,2C): one 16B store
          *(v8h*)&outQK[((size_t)bi * N + col) * 1024 + rowbase] = pk;
        } else {                // v -> natural (B,C,T)
#pragma unroll
          for (int r = 0; r < 8; ++r)
            outV[((size_t)bi * 512 + (rowbase - 1024 + r)) * N + col] = pk[r];
        }
      } else {
#pragma unroll
        for (int r = 0; r < 8; ++r) {
          int row = rowbase + r;
          size_t idx = ((size_t)bi * M + row) * N + col;
          outF[idx] = resid[idx] + acc[mt][nt][r] + bias[row];
        }
      }
    }
  }
}

// ---------------------------------------------------------------------------
// Flash attention: block = (b, head, 64 query rows), 4 waves x 16 rows.
// Q fragments hoisted out of the s-loop; K/V tiles staged async to LDS.
// ---------------------------------------------------------------------------
#define ALD 72   // LDS leading dim (halves) for Qt/Kt/Vt/Pt

__global__ __launch_bounds__(128)
void attn_kernel(const _Float16* __restrict__ qkT, const _Float16* __restrict__ vN,
                 _Float16* __restrict__ oT, int C, int T, int H) {
  const int hd = 64;
  int t0 = blockIdx.x * 64;
  int hh = blockIdx.y;
  int bi = blockIdx.z;
  const int CQK = 1024;                                   // 2*C
  const _Float16* qk = qkT + (size_t)bi * T * CQK;
  const _Float16* vb = vN + ((size_t)bi * 512 + hh * hd) * T;

  __shared__ __align__(16) _Float16 Qt[64 * ALD];   // [t][d] (A op, pre-scaled)
  __shared__ __align__(16) _Float16 Kt[64 * ALD];   // [s][d] (B op, k=d contig)
  __shared__ __align__(16) _Float16 Vt[64 * ALD];   // [d][s] (A op, k=s contig)
  __shared__ __align__(16) _Float16 Pt[4][16 * ALD];// per-wave [t][s] (B op)
  __shared__ float sAlpha[4][16];
  __shared__ float sL[4][16];

  int tid = threadIdx.x, wave = tid >> 5, lane = tid & 31;
  int n_l = lane & 15, hf = (lane >> 4) & 1;

  // stage Q once (scale already folded in by the QKV GEMM epilogue)
  for (int i = tid; i < 512; i += 128) {
    int tl = i >> 3, dc = (i & 7) * 8;
    cp16_async(&Qt[tl * ALD + dc], &qk[(size_t)(t0 + tl) * CQK + hh * hd + dc]);
  }
  async_wait0();
  __syncthreads();
  // loop-invariant Q A-fragments (wave's 16 t-rows, K = d in two 32-steps)
  v16h qa0 = load_a_frag(&Qt[(wave * 16) * ALD + 0], ALD);
  v16h qa1 = load_a_frag(&Qt[(wave * 16) * ALD + 32], ALD);

  v8f O[4] = {};
  float mrow[8], lrow[8];
#pragma unroll
  for (int r = 0; r < 8; ++r) { mrow[r] = -3.0e38f; lrow[r] = 0.f; }

  for (int s0 = 0; s0 < T; s0 += 64) {
    __syncthreads();                 // previous K/V/P consumers done
    for (int i = tid; i < 512; i += 128) {
      int sl = i >> 3, dc = (i & 7) * 8;
      cp16_async(&Kt[sl * ALD + dc],
                 &qk[(size_t)(s0 + sl) * CQK + 512 + hh * hd + dc]);
    }
    for (int i = tid; i < 512; i += 128) {
      int d = i >> 3, sc = (i & 7) * 8;
      cp16_async(&Vt[d * ALD + sc], &vb[(size_t)d * T + s0 + sc]);
    }
    async_wait0();
    __syncthreads();

    // S = Q^T K : wave's 16 t-rows x 64 s-cols
    v8f S[4] = {};
#pragma unroll
    for (int nt = 0; nt < 4; ++nt) {
      v16h b0 = load_b_frag_nk(&Kt[(nt * 16) * ALD + 0], ALD);
      S[nt] = wmma_f16(qa0, b0, S[nt]);
      v16h b1 = load_b_frag_nk(&Kt[(nt * 16) * ALD + 32], ALD);
      S[nt] = wmma_f16(qa1, b1, S[nt]);
    }

    // online softmax per row (xor<16 shuffles stay within each 16-lane half)
    float alpha[8];
#pragma unroll
    for (int r = 0; r < 8; ++r) {
      float mx = S[0][r];
#pragma unroll
      for (int nt = 1; nt < 4; ++nt) mx = fmaxf(mx, S[nt][r]);
#pragma unroll
      for (int off = 1; off < 16; off <<= 1)
        mx = fmaxf(mx, __shfl_xor(mx, off, 32));
      float mnew = fmaxf(mrow[r], mx);
      alpha[r] = __expf(mrow[r] - mnew);
      float psum = 0.f;
#pragma unroll
      for (int nt = 0; nt < 4; ++nt) {
        float p = __expf(S[nt][r] - mnew);
        S[nt][r] = p;
        psum += p;
      }
#pragma unroll
      for (int off = 1; off < 16; off <<= 1)
        psum += __shfl_xor(psum, off, 32);
      lrow[r] = lrow[r] * alpha[r] + psum;
      mrow[r] = mnew;
    }

    // P -> LDS as [t][s] so it re-feeds as a contiguous B operand
#pragma unroll
    for (int nt = 0; nt < 4; ++nt)
#pragma unroll
      for (int r = 0; r < 8; ++r)
        Pt[wave][(r + hf * 8) * ALD + nt * 16 + n_l] = (_Float16)S[nt][r];
    if (n_l == 0)
#pragma unroll
      for (int r = 0; r < 8; ++r) sAlpha[wave][r + hf * 8] = alpha[r];
    __syncthreads();

    // rescale O (alpha indexed by output column t = lane&15) and O += V P^T
    float at = sAlpha[wave][n_l];
#pragma unroll
    for (int dt = 0; dt < 4; ++dt)
#pragma unroll
      for (int r = 0; r < 8; ++r) O[dt][r] *= at;
#pragma unroll
    for (int kd = 0; kd < 64; kd += 32) {
      v16h b = load_b_frag_nk(&Pt[wave][kd], ALD);
#pragma unroll
      for (int dt = 0; dt < 4; ++dt) {
        v16h a = load_a_frag(&Vt[(dt * 16) * ALD + kd], ALD);
        O[dt] = wmma_f16(a, b, O[dt]);
      }
    }
  }

  // finalize: divide by row sum, write oT (B,T,C) -- 8 consecutive d per lane
  if (n_l == 0)
#pragma unroll
    for (int r = 0; r < 8; ++r) sL[wave][r + hf * 8] = lrow[r];
  __syncthreads();
  float linv = 1.f / sL[wave][n_l];
  int t = t0 + wave * 16 + n_l;
#pragma unroll
  for (int dt = 0; dt < 4; ++dt) {
    v8h pk;
#pragma unroll
    for (int r = 0; r < 8; ++r) pk[r] = (_Float16)(O[dt][r] * linv);
    *(v8h*)&oT[((size_t)bi * T + t) * 512 + hh * hd + dt * 16 + hf * 8] = pk;
  }
}

// ---------------------------------------------------------------------------
extern "C" void kernel_launch(void* const* d_in, const int* in_sizes, int n_in,
                              void* d_out, int out_size, void* d_ws, size_t ws_size,
                              hipStream_t stream) {
  (void)in_sizes; (void)n_in; (void)out_size; (void)ws_size;
  const float* x      = (const float*)d_in[0];
  const float* gn_w   = (const float*)d_in[1];
  const float* gn_b   = (const float*)d_in[2];
  const float* qkv_w  = (const float*)d_in[3];
  const float* qkv_b  = (const float*)d_in[4];
  const float* proj_w = (const float*)d_in[5];
  const float* proj_b = (const float*)d_in[6];
  float* out = (float*)d_out;

  const int B = 4, C = 512, T = 2048, H = 8, G = 8;

  // workspace (f16): hT | qkv_w | proj_w | qkT | vN | oT   (~44 MB)
  _Float16* hT     = (_Float16*)d_ws;
  _Float16* qkvwF  = hT + (size_t)B * T * C;
  _Float16* projwF = qkvwF + (size_t)3 * C * C;
  _Float16* qkT    = projwF + (size_t)C * C;        // (B, T, 2C)
  _Float16* vNat   = qkT + (size_t)B * T * 2 * C;   // (B, C, T)
  _Float16* oT     = vNat + (size_t)B * C * T;      // (B, T, C)

  gn_kernel<<<dim3(B * G), 256, 0, stream>>>(x, gn_w, gn_b, hT, C, T, G);
  cvt_kernel<<<dim3((3 * C * C + 255) / 256), 256, 0, stream>>>(qkv_w, qkvwF, 3 * C * C);
  cvt_kernel<<<dim3((C * C + 255) / 256), 256, 0, stream>>>(proj_w, projwF, C * C);

  gemm_kernel<0><<<dim3(T / GBN, (3 * C) / GBM, B), 128, 0, stream>>>(
      qkvwF, hT, qkv_b, nullptr, qkT, vNat, nullptr, 3 * C, C, T);

  attn_kernel<<<dim3(T / 64, H, B), 128, 0, stream>>>(qkT, vNat, oT, C, T, H);

  gemm_kernel<1><<<dim3(T / GBN, C / GBM, B), 128, 0, stream>>>(
      projwF, oT, proj_b, x, nullptr, nullptr, out, C, C, T);
}
